// GATLayerSingleBatch_19172734010027
// MI455X (gfx1250) — matrix operand order repaired and checked by reading, not compiled, not run on hardware
//
#include <hip/hip_runtime.h>

// GAT layer, fused flash-attention style, f16 WMMA (16x16x32) with f32 accum.
// N=4096 nodes, F=128 in-feat, H=4 heads, O=D=64.
//  - adjacency bit-packed by a wave32 __ballot pass (67MB -> 2MB working set)
//  - 1/sqrt(D) folded into Q (LeakyReLU is positively homogeneous)
//  - softmax denominator via a ones-column WMMA tile (register-constant B)
//  - P transposed C->A layout through a wave-private LDS tile

typedef __attribute__((ext_vector_type(16))) _Float16 v16h;
typedef __attribute__((ext_vector_type(8)))  _Float16 v8h;
typedef __attribute__((ext_vector_type(8)))  float    v8f;

#define NN 4096
#define FF 128
#define HH 4
#define OO 64
#define DD 64
#define NW (NN / 32)   // 128 adjacency words per row

__device__ __forceinline__ v8f wmma_f16(v16h a, v16h b, v8f c) {
  // 8 args: (neg_a, A, neg_b, B, c_mod, C, reuse_a, reuse_b)
  return __builtin_amdgcn_wmma_f32_16x16x32_f16(false, a, false, b, (short)0, c,
                                                false, false);
}

// ---------------------------------------------------------------- kernel 0a
// Per-row: convert X to f16 (re-read 12x by projections) + nonzero row mask.
__global__ void __launch_bounds__(256) rowcvt_kernel(const float* __restrict__ x,
                                                     _Float16* __restrict__ xh,
                                                     float* __restrict__ rowmask) {
  const int r = blockIdx.x * blockDim.x + threadIdx.x;
  if (r >= NN) return;
  const float4* p = (const float4*)(x + (size_t)r * FF);
  _Float16* dst = xh + (size_t)r * FF;
  float any = 0.f;
#pragma unroll 4
  for (int i = 0; i < FF / 4; ++i) {
    float4 v = p[i];
    any += (v.x != 0.f ? 1.f : 0.f) + (v.y != 0.f ? 1.f : 0.f) +
           (v.z != 0.f ? 1.f : 0.f) + (v.w != 0.f ? 1.f : 0.f);
    dst[i * 4 + 0] = (_Float16)v.x;
    dst[i * 4 + 1] = (_Float16)v.y;
    dst[i * 4 + 2] = (_Float16)v.z;
    dst[i * 4 + 3] = (_Float16)v.w;
  }
  rowmask[r] = (any > 0.f) ? 1.f : 0.f;
}

// ---------------------------------------------------------------- kernel 0b
// Transpose + convert the three weight tensors: wbufT[(w*H+h)*64+o][128] f16.
__global__ void __launch_bounds__(256) wtrans_kernel(const float* __restrict__ wa,
                                                     const float* __restrict__ wa2,
                                                     const float* __restrict__ wk,
                                                     _Float16* __restrict__ wbufT) {
  const int t = blockIdx.x * blockDim.x + threadIdx.x;  // 3*H*64 = 768 rows
  if (t >= 3 * HH * 64) return;
  const int w = t / (HH * 64);
  const int h = (t / 64) % HH;
  const int o = t % 64;
  const float* src = (w == 0 ? wa : (w == 1 ? wa2 : wk)) +
                     (size_t)h * FF * 64 + o;
  _Float16* dst = wbufT + (size_t)t * FF;
#pragma unroll 8
  for (int f = 0; f < FF; ++f) dst[f] = (_Float16)src[(size_t)f * 64];
}

// ---------------------------------------------------------------- kernel 0c
// Bit-pack the binary adjacency: 32 f32 -> one u32 word via wave32 ballot.
// Each wave packs 64 consecutive words (2048 floats), fully coalesced.
__global__ void __launch_bounds__(256) packadj_kernel(const float* __restrict__ adj,
                                                      unsigned* __restrict__ adjbits) {
  const int wave = threadIdx.x >> 5;
  const int lane = threadIdx.x & 31;
  const int wid = blockIdx.x * 8 + wave;          // 8192 waves
  const size_t base = (size_t)wid * 64;           // first word for this wave
#pragma unroll 4
  for (int i = 0; i < 64; ++i) {
    const float v = adj[(base + i) * 32 + lane];
    const unsigned m = (unsigned)__ballot(v != 0.f);
    if (lane == 0) adjbits[base + i] = m;
  }
}

// ---------------------------------------------------------------- kernel 1
// One wave per (head, 16-row tile, which-of-{Q,K,V}). C = X(16x128) * W(128x64).
// All fragment loads are contiguous 16B/32B vector loads. Q scaled by 1/8.
__global__ void __launch_bounds__(256) proj_kernel(
    const _Float16* __restrict__ xh,    // [N][F] f16
    const _Float16* __restrict__ wbufT, // [(w*H+h)*64+o][F] f16
    _Float16* __restrict__ qbuf,        // [H][N][D]  (pre-scaled by 0.125)
    _Float16* __restrict__ kbuf,        // [H][N][D]
    _Float16* __restrict__ vbufT)       // [H][O][N]  (transposed feats)
{
  const int wave = threadIdx.x >> 5;
  const int task = blockIdx.x * 8 + wave;        // 3072 tasks total
  const int which = task % 3;
  const int mt = (task / 3) % (NN / 16);
  const int h = task / (3 * (NN / 16));
  const int lane = threadIdx.x & 31;
  const int lcol = lane & 15;
  const bool hihalf = lane >= 16;
  const int m16 = mt * 16;

  // A fragments from xh. contiguous k order used here (B matches):
  // lanes 0-15: halves 0..15 = K+0..15 ; hi-half +16.
  v16h afrag[4];
  {
    const int row = m16 + lcol;
    const _Float16* xr = xh + (size_t)row * FF;
    const int kb = hihalf ? 16 : 0;
#pragma unroll
    for (int kc = 0; kc < 4; ++kc) {
      const int k0 = kc * 32 + kb;
      v8h lo = *(const v8h*)(xr + k0);
      v8h hg = *(const v8h*)(xr + k0 + 8);
      v16h a;
#pragma unroll
      for (int j = 0; j < 8; ++j) { a[j] = lo[j]; a[j + 8] = hg[j]; }
      afrag[kc] = a;
    }
  }

  const _Float16* WT = wbufT + (size_t)(which * HH + h) * 64 * FF;
  const int kbB = hihalf ? 16 : 0;   // B 32x16: lanes 0-15 K=0..15, hi K=16..31
  const int roff = hihalf ? 8 : 0;   // C layout row offset
  const float oscale = (which == 0) ? 0.125f : 1.0f;  // fold 1/sqrt(D) into Q

#pragma unroll
  for (int nt = 0; nt < 4; ++nt) {
    const int col = nt * 16 + lcol;
    v8f acc = {};
#pragma unroll
    for (int kc = 0; kc < 4; ++kc) {
      v16h b = *(const v16h*)(WT + (size_t)col * FF + kc * 32 + kbB);
      acc = wmma_f16(afrag[kc], b, acc);
    }
    if (which == 2) {
      // feats transposed: vbufT[h][o][node]; per-lane 8 consecutive nodes.
      _Float16* dst = vbufT + (size_t)h * OO * NN + (size_t)col * NN + m16 + roff;
#pragma unroll
      for (int r = 0; r < 8; ++r) dst[r] = (_Float16)acc[r];
    } else {
      _Float16* dst = (which == 0 ? qbuf : kbuf) + (size_t)h * NN * DD +
                      (size_t)(m16 + roff) * DD + col;
#pragma unroll
      for (int r = 0; r < 8; ++r) dst[(size_t)r * DD] = (_Float16)(acc[r] * oscale);
    }
  }
}

// ---------------------------------------------------------------- kernel 2
// One wave per (head, 16-row tile): fused S = Q K^T, masked-leaky softmax
// (online), O += P V. Wave-private LDS tile transposes P (C layout->A layout);
// same-wave LDS ops are in-order on CDNA5, so only a compiler fence is needed.
__global__ void __launch_bounds__(256) gat_attn_kernel(
    const unsigned* __restrict__ adjbits, // [N][N/32] packed 0/1
    const _Float16* __restrict__ qbuf,    // [H][N][D] (x 0.125)
    const _Float16* __restrict__ kbuf,    // [H][N][D]
    const _Float16* __restrict__ vbufT,   // [H][O][N]
    const float* __restrict__ rowmask,    // [N]
    float* __restrict__ out)              // [N][H*O]
{
  __shared__ __align__(16) _Float16 plds[8][16][40];  // 40-half row pitch

  const int wave = threadIdx.x >> 5;
  const int task = blockIdx.x * 8 + wave;   // 1024 tasks total
  const int mt = task % (NN / 16);
  const int h = task / (NN / 16);
  const int lane = threadIdx.x & 31;
  const int lcol = lane & 15;
  const bool hihalf = lane >= 16;
  const int roff = hihalf ? 8 : 0;
  const int m16 = mt * 16;

  // Q as two A fragments (D=64 -> two K=32 chunks), contiguous 16B loads.
  v16h qf[2];
  {
    const int node = m16 + lcol;
    const _Float16* qr = qbuf + (size_t)h * NN * DD + (size_t)node * DD;
    const int kb = hihalf ? 16 : 0;
#pragma unroll
    for (int kc = 0; kc < 2; ++kc) {
      const int k0 = kc * 32 + kb;
      v8h lo = *(const v8h*)(qr + k0);
      v8h hg = *(const v8h*)(qr + k0 + 8);
      v16h a;
#pragma unroll
      for (int j = 0; j < 8; ++j) { a[j] = lo[j]; a[j + 8] = hg[j]; }
      qf[kc] = a;
    }
  }

  // Ones-column B fragment for the row-sum tile: B[k][n] = (n==0).
  v16h bones;
  {
    const _Float16 v = (lcol == 0) ? (_Float16)1.0f : (_Float16)0.0f;
#pragma unroll
    for (int j = 0; j < 16; ++j) bones[j] = v;
  }

  v8f accO[5] = {};                // 4 output tiles + 1 row-sum tile
  float mstate[8];
#pragma unroll
  for (int r = 0; r < 8; ++r) mstate[r] = -1e30f;

  const int kbB = hihalf ? 16 : 0;
  const _Float16* kbase = kbuf + (size_t)h * NN * DD;
  const _Float16* vbase = vbufT + (size_t)h * OO * NN;
  const unsigned* abase = adjbits + (size_t)(m16 + roff) * NW;

  for (int jc = 0; jc < NN; jc += 32) {
    // ---- S tiles: cols jc..jc+15 (s0) and jc+16..jc+31 (s1)
    v8f s0 = {}, s1 = {};
    {
      const _Float16* kr0 = kbase + (size_t)(jc + lcol) * DD;   // B col=node
      const _Float16* kr1 = kr0 + (size_t)16 * DD;
      v16h b00 = *(const v16h*)(kr0 + kbB);        // d 0..31 chunk
      v16h b01 = *(const v16h*)(kr0 + 32 + kbB);   // d 32..63 chunk
      v16h b10 = *(const v16h*)(kr1 + kbB);
      v16h b11 = *(const v16h*)(kr1 + 32 + kbB);
      s0 = wmma_f16(qf[0], b00, s0);
      s0 = wmma_f16(qf[1], b01, s0);
      s1 = wmma_f16(qf[0], b10, s1);
      s1 = wmma_f16(qf[1], b11, s1);
    }
    const int jw = jc >> 5;   // one 32-bit word covers both 16-col tiles

    // ---- adjacency bits + LeakyReLU (scale pre-folded into Q), row-max
    float lg0[8], lg1[8], tmax[8];
#pragma unroll
    for (int r = 0; r < 8; ++r) {
      const unsigned w = abase[(size_t)r * NW + jw];
      float v0 = s0[r]; v0 = (v0 >= 0.f) ? v0 : 0.2f * v0;
      float v1 = s1[r]; v1 = (v1 >= 0.f) ? v1 : 0.2f * v1;
      lg0[r] = ((w >> lcol) & 1u) ? v0 : -1e30f;
      lg1[r] = ((w >> (lcol + 16)) & 1u) ? v1 : -1e30f;
      float t = fmaxf(lg0[r], lg1[r]);
      t = fmaxf(t, __shfl_xor(t, 1));
      t = fmaxf(t, __shfl_xor(t, 2));
      t = fmaxf(t, __shfl_xor(t, 4));
      t = fmaxf(t, __shfl_xor(t, 8));   // reduce within 16-lane half
      tmax[r] = t;
    }

    // ---- online softmax update + write P into per-wave LDS tile
#pragma unroll
    for (int r = 0; r < 8; ++r) {
      const float mnew = fmaxf(mstate[r], tmax[r]);
      const float c = __expf(mstate[r] - mnew);
      mstate[r] = mnew;
      const float p0 = (lg0[r] > -1e29f) ? __expf(lg0[r] - mnew) : 0.f;
      const float p1 = (lg1[r] > -1e29f) ? __expf(lg1[r] - mnew) : 0.f;
#pragma unroll
      for (int nt = 0; nt < 5; ++nt) accO[nt][r] *= c;
      plds[wave][roff + r][lcol]      = (_Float16)p0;
      plds[wave][roff + r][16 + lcol] = (_Float16)p1;
    }
    __builtin_amdgcn_wave_barrier();   // ordering fence; LDS is wave-private

    // ---- P (C layout) -> A fragment via LDS
    v16h ap;
    {
      const _Float16* pr = &plds[wave][lcol][0];
      const int cb = hihalf ? 16 : 0;
      v8h lo = *(const v8h*)(pr + cb);
      v8h hg = *(const v8h*)(pr + 8 + cb);
#pragma unroll
      for (int j = 0; j < 8; ++j) { ap[j] = lo[j]; ap[j + 8] = hg[j]; }
    }
    __builtin_amdgcn_wave_barrier();

    // ---- accO += P(16x32) * V(32x64); V transposed so B loads are 32B runs
#pragma unroll
    for (int nt = 0; nt < 4; ++nt) {
      const int o = nt * 16 + lcol;
      v16h bv = *(const v16h*)(vbase + (size_t)o * NN + jc + kbB);
      accO[nt] = wmma_f16(ap, bv, accO[nt]);
    }
    // row-sum tile: denominator of the softmax, rescaled with the rest.
    accO[4] = wmma_f16(ap, bones, accO[4]);
  }

  // ---- epilogue: normalize, ReLU, row mask, store f32 [N][H*O]
#pragma unroll
  for (int r = 0; r < 8; ++r) {
    const int node = m16 + roff + r;
    const float rm = rowmask[node];
    // row-sum lives in column 0 of tile 4 (lanes 0 and 16); broadcast.
    const float rs = __shfl(accO[4][r], hihalf ? 16 : 0);
    const float inv = (rs > 0.f) ? (1.f / rs) : 0.f;
    float* op = out + (size_t)node * (HH * OO) + h * OO;
#pragma unroll
    for (int nt = 0; nt < 4; ++nt) {
      float v = accO[nt][r] * inv;
      v = fmaxf(v, 0.f) * rm;
      op[nt * 16 + lcol] = v;
    }
  }
}

// ---------------------------------------------------------------- launcher
extern "C" void kernel_launch(void* const* d_in, const int* in_sizes, int n_in,
                              void* d_out, int out_size, void* d_ws, size_t ws_size,
                              hipStream_t stream) {
  const float* x   = (const float*)d_in[0];  // h [1,N,F]
  const float* adj = (const float*)d_in[1];  // a [1,N,N]
  const float* wk  = (const float*)d_in[2];  // kernel [H,F,O]
  const float* wa  = (const float*)d_in[3];  // attn_kernel [H,F,D]
  const float* wa2 = (const float*)d_in[4];  // attn_kernel2 [H,F,D]
  float* out = (float*)d_out;                // [1,N,H*O]

  char* ws = (char*)d_ws;
  size_t off = 0;
  const size_t szHND = (size_t)HH * NN * DD * sizeof(_Float16);  // 2 MB
  _Float16* qbuf  = (_Float16*)(ws + off); off += szHND;
  _Float16* kbuf  = (_Float16*)(ws + off); off += szHND;
  _Float16* vbufT = (_Float16*)(ws + off); off += szHND;
  _Float16* xh    = (_Float16*)(ws + off); off += (size_t)NN * FF * sizeof(_Float16);
  _Float16* wbufT = (_Float16*)(ws + off); off += (size_t)3 * HH * 64 * FF * sizeof(_Float16);
  float*    rmask = (float*)(ws + off);    off += (size_t)NN * sizeof(float);
  unsigned* abits = (unsigned*)(ws + off); off += (size_t)NN * NW * sizeof(unsigned);

  rowcvt_kernel<<<NN / 256, 256, 0, stream>>>(x, xh, rmask);
  wtrans_kernel<<<3, 256, 0, stream>>>(wa, wa2, wk, wbufT);
  packadj_kernel<<<(NN * NW) / (64 * 8), 256, 0, stream>>>(adj, abits);
  proj_kernel<<<(HH * (NN / 16) * 3) / 8, 256, 0, stream>>>(
      xh, wbufT, qbuf, kbuf, vbufT);
  gat_attn_kernel<<<(HH * (NN / 16)) / 8, 256, 0, stream>>>(
      abits, qbuf, kbuf, vbufT, rmask, out);
}
